// GraphAttnMsgPassingEGNNLayer_55688545960336
// MI455X (gfx1250) — compile-verified, matrix operand order checked by reading, not run on hardware
//
#include <hip/hip_runtime.h>

// ---------------- WMMA types/helpers (gfx1250, wave32) ----------------
typedef __attribute__((ext_vector_type(16))) _Float16     v16h;
typedef __attribute__((ext_vector_type(8)))  float        v8f;
typedef __attribute__((ext_vector_type(4)))  unsigned int u32x4; // trivial 16B vector

union HV16 { v16h v; u32x4 q[2]; };

__device__ __forceinline__ v8f wmma32(v16h a, v16h b, v8f c) {
  return __builtin_amdgcn_wmma_f32_16x16x32_f16(false, a, false, b, (short)0, c,
                                                false, false);
}

// A fragment: 16x32 f16 tile from row-major buffer (ld in halves).
// ISA layout: lane l holds row l%16, group g=l/16; half i -> K = 16*(i/8) + 8g + (i%8)
// => two contiguous 16B loads per lane.
__device__ __forceinline__ v16h frag_a(const _Float16* src, int ld) {
  const int lane = threadIdx.x & 31;
  const int row  = lane & 15;
  const int g    = lane >> 4;
  const _Float16* p = src + row * ld + g * 8;
  HV16 t;
  t.q[0] = *(const u32x4*)(p);
  t.q[1] = *(const u32x4*)(p + 16);
  return t.v;
}

// B fragment from pre-swizzled weights: each (ktile,ntile) block is 32 lanes x 16 halves
// stored contiguously; lane just loads its 32 bytes.
__device__ __forceinline__ v16h frag_b_sw(const _Float16* p) {
  const int lane = threadIdx.x & 31;
  HV16 t;
  const u32x4* q = (const u32x4*)(p + (size_t)lane * 16);
  t.q[0] = q[0];
  t.q[1] = q[1];
  return t.v;
}

// LayerNorm(+affine)+ReLU over one 128-wide row (f32 in LDS -> f16 in LDS)
__device__ __forceinline__ void ln_relu_row(const float* row, const float* gamma,
                                            const float* beta, _Float16* outrow) {
  float m = 0.f;
  for (int c = 0; c < 128; ++c) m += row[c];
  m *= (1.0f / 128.0f);
  float v = 0.f;
  for (int c = 0; c < 128; ++c) { float t = row[c] - m; v += t * t; }
  v *= (1.0f / 128.0f);
  const float rinv = rsqrtf(v + 1e-5f);
  for (int c = 0; c < 128; ++c) {
    float t = (row[c] - m) * rinv * gamma[c] + beta[c];
    outrow[c] = (_Float16)fmaxf(t, 0.0f);
  }
}

constexpr int TILE_E = 64;
constexpr int TILE_N = 64;

// ---------------- prep kernels ----------------
__global__ void cast_f16_kernel(const float* __restrict__ in, _Float16* __restrict__ out, int n) {
  for (int i = blockIdx.x * blockDim.x + threadIdx.x; i < n; i += gridDim.x * blockDim.x)
    out[i] = (_Float16)in[i];
}

// Swizzle a [K x Nn] row-major f32 weight into WMMA-B fragment order (f16).
__global__ void swizzle_b_kernel(const float* __restrict__ w, _Float16* __restrict__ o,
                                 int K, int Nn) {
  int idx = blockIdx.x * blockDim.x + threadIdx.x;
  if (idx >= K * Nn) return;
  int k = idx / Nn, n = idx % Nn;
  int kt = k >> 5, r = k & 31;
  int i  = ((r >> 4) << 3) | (r & 7);
  int g  = (r >> 3) & 1;
  int lane = (g << 4) | (n & 15);
  int nt = n >> 4;
  int ntiles = Nn >> 4;
  size_t dst = (((size_t)kt * ntiles + nt) * 32 + lane) * 16 + i;
  o[dst] = (_Float16)w[idx];
}

// ---------------- edge kernel: features, kv/q GEMMs, logits, v, atomic max ----------------
__global__ __launch_bounds__(128) void edge_main_kernel(
    const float* __restrict__ x, const float* __restrict__ edge_attr,
    const int* __restrict__ ei, int E,
    const _Float16* __restrict__ hf,
    const _Float16* __restrict__ kv_w1h, const float* __restrict__ kv_b1,
    const float* __restrict__ kv_g, const float* __restrict__ kv_be,
    const _Float16* __restrict__ kv_w2h, const float* __restrict__ kv_b2,
    const _Float16* __restrict__ q_wh, const float* __restrict__ q_b,
    const _Float16* __restrict__ ab_w1h, const float* __restrict__ ab_b1,
    const float* __restrict__ ab_g, const float* __restrict__ ab_be,
    const float* __restrict__ ab_w2, const float* __restrict__ ab_b2,
    _Float16* __restrict__ vout, float* __restrict__ exbuf, unsigned* __restrict__ nmax) {
  __shared__ __align__(16) _Float16 sA[TILE_E * 192];   // [h_src | edge_feat]
  __shared__ __align__(16) _Float16 sQ[TILE_E * 128];   // h_dst
  __shared__ __align__(16) float    sY32[TILE_E * 128]; // GEMM f32 staging
  __shared__ __align__(16) _Float16 sYh[TILE_E * 128];  // LN+relu f16
  __shared__ float sLog[TILE_E * 8];
  __shared__ int sSrc[TILE_E], sDst[TILE_E], sVal[TILE_E];

  const int tid  = threadIdx.x;
  const int wave = tid >> 5;
  const int lane = tid & 31;
  const int base = blockIdx.x * TILE_E;

  // per-edge meta + RBF x edge_attr outer product (cols 128..191 of sA)
  if (tid < TILE_E) {
    int e = base + tid;
    int valid = (e < E);
    int s = valid ? ei[e] : 0;
    int d = valid ? ei[E + e] : 0;
    sSrc[tid] = s; sDst[tid] = d; sVal[tid] = valid;
    float rx = 0.f, ry = 0.f, rz = 0.f, dist = 0.f;
    float ea0 = 0.f, ea1 = 0.f, ea2 = 0.f, ea3 = 0.f;
    if (valid) {
      rx = x[d * 3 + 0] - x[s * 3 + 0];
      ry = x[d * 3 + 1] - x[s * 3 + 1];
      rz = x[d * 3 + 2] - x[s * 3 + 2];
      dist = sqrtf(rx * rx + ry * ry + rz * rz + 1e-8f);
      ea0 = edge_attr[(size_t)e * 4 + 0];
      ea1 = edge_attr[(size_t)e * 4 + 1];
      ea2 = edge_attr[(size_t)e * 4 + 2];
      ea3 = edge_attr[(size_t)e * 4 + 3];
    }
    const float step  = 10.0f / 15.0f;
    const float coeff = -1.125f; // -0.5/step^2
#pragma unroll
    for (int r = 0; r < 16; ++r) {
      float dd = dist - step * (float)r;
      float df = valid ? __expf(coeff * dd * dd) : 0.f;
      _Float16* q = sA + tid * 192 + 128 + r * 4;
      q[0] = (_Float16)(df * ea0);
      q[1] = (_Float16)(df * ea1);
      q[2] = (_Float16)(df * ea2);
      q[3] = (_Float16)(df * ea3);
    }
  }
  __syncthreads();

  // gather h[src] -> sA cols 0..127, h[dst] -> sQ (vectorized 16B)
  {
    int eL = tid >> 1;
    int off = (tid & 1) << 6; // 0 or 64 halves
    int s = sSrc[eL], d = sDst[eL];
    bool valid = sVal[eL] != 0;
    const u32x4* ps = (const u32x4*)(hf + (size_t)s * 128 + off);
    const u32x4* pd = (const u32x4*)(hf + (size_t)d * 128 + off);
    u32x4* qa = (u32x4*)(sA + eL * 192 + off);
    u32x4* qq = (u32x4*)(sQ + eL * 128 + off);
    u32x4 z = {0u, 0u, 0u, 0u};
#pragma unroll
    for (int i = 0; i < 8; ++i) {
      qa[i] = valid ? ps[i] : z;
      qq[i] = valid ? pd[i] : z;
    }
  }
  __syncthreads();

  // GEMM1: kv_in[64,192] @ kv_w1[192,128] -> sY32 (+bias)
  {
    const _Float16* Abase = sA + wave * 16 * 192;
    const int g = lane >> 4, cl = lane & 15;
    v16h a[6];
#pragma unroll
    for (int kk = 0; kk < 6; ++kk) a[kk] = frag_a(Abase + kk * 32, 192);
    for (int nt = 0; nt < 8; ++nt) {
      v8f c = {};
#pragma unroll
      for (int kk = 0; kk < 6; ++kk)
        c = wmma32(a[kk], frag_b_sw(kv_w1h + (size_t)(kk * 8 + nt) * 512), c);
      int col = nt * 16 + cl;
      float b = kv_b1[col];
#pragma unroll
      for (int r = 0; r < 8; ++r)
        sY32[(wave * 16 + r + 8 * g) * 128 + col] = c[r] + b;
    }
  }
  __syncthreads();
  if (tid < TILE_E) ln_relu_row(sY32 + tid * 128, kv_g, kv_be, sYh + tid * 128);
  __syncthreads();

  // GEMM2 (k half of kv_w2) + q GEMM, one head per 16-wide N-tile -> logits
  {
    const _Float16* Aq = sQ + wave * 16 * 128;
    const _Float16* Ak = sYh + wave * 16 * 128;
    const int g = lane >> 4, cl = lane & 15;
    const float scale = 0.25f; // 1/sqrt(16)
    v16h aq[4], ak[4];
#pragma unroll
    for (int kk = 0; kk < 4; ++kk) {
      aq[kk] = frag_a(Aq + kk * 32, 128);
      ak[kk] = frag_a(Ak + kk * 32, 128);
    }
    for (int j = 0; j < 8; ++j) {
      v8f cq = {}, ck = {};
#pragma unroll
      for (int kk = 0; kk < 4; ++kk) {
        cq = wmma32(aq[kk], frag_b_sw(q_wh + (size_t)(kk * 8 + j) * 512), cq);
        ck = wmma32(ak[kk], frag_b_sw(kv_w2h + (size_t)(kk * 16 + j) * 512), ck);
      }
      int col = j * 16 + cl;
      float qb = q_b[col], kb = kv_b2[col];
      float prod[8];
#pragma unroll
      for (int r = 0; r < 8; ++r) prod[r] = (cq[r] + qb) * (ck[r] + kb);
#pragma unroll
      for (int m2 = 1; m2 <= 8; m2 <<= 1)
#pragma unroll
        for (int r = 0; r < 8; ++r) prod[r] += __shfl_xor(prod[r], m2, 32);
      if (cl == 0) {
#pragma unroll
        for (int r = 0; r < 8; ++r)
          sLog[(wave * 16 + r + 8 * g) * 8 + j] = prod[r] * scale;
      }
    }
    // v half of kv_w2 -> global f16 v buffer
    for (int j = 0; j < 8; ++j) {
      v8f cv = {};
#pragma unroll
      for (int kk = 0; kk < 4; ++kk)
        cv = wmma32(ak[kk], frag_b_sw(kv_w2h + (size_t)(kk * 16 + 8 + j) * 512), cv);
      int col = j * 16 + cl;
      float vb = kv_b2[128 + col];
#pragma unroll
      for (int r = 0; r < 8; ++r) {
        int erow = wave * 16 + r + 8 * g;
        if (sVal[erow]) vout[(size_t)(base + erow) * 128 + col] = (_Float16)(cv[r] + vb);
      }
    }
  }
  __syncthreads();

  // attention-bias MLP: edge_feat @ ab_w1 -> LN/relu (sY32/sYh are free now)
  {
    const _Float16* Ae = sA + wave * 16 * 192 + 128;
    const int g = lane >> 4, cl = lane & 15;
    v16h a[2];
#pragma unroll
    for (int kk = 0; kk < 2; ++kk) a[kk] = frag_a(Ae + kk * 32, 192);
    for (int nt = 0; nt < 8; ++nt) {
      v8f c = {};
#pragma unroll
      for (int kk = 0; kk < 2; ++kk)
        c = wmma32(a[kk], frag_b_sw(ab_w1h + (size_t)(kk * 8 + nt) * 512), c);
      int col = nt * 16 + cl;
      float b = ab_b1[col];
#pragma unroll
      for (int r = 0; r < 8; ++r)
        sY32[(wave * 16 + r + 8 * g) * 128 + col] = c[r] + b;
    }
  }
  __syncthreads();
  if (tid < TILE_E) ln_relu_row(sY32 + tid * 128, ab_g, ab_be, sYh + tid * 128);
  __syncthreads();

  // bias @ ab_w2 (128->8 VALU dot), finalize logits, atomic segment-max
  for (int t = tid; t < TILE_E * 8; t += 128) {
    int e = t >> 3, j = t & 7;
    float acc = ab_b2[j];
    const _Float16* row = sYh + e * 128;
    for (int c = 0; c < 128; ++c) acc += (float)row[c] * ab_w2[c * 8 + j];
    float lg = sLog[t] + acc;
    if (sVal[e]) {
      int ge = base + e;
      exbuf[(size_t)ge * 8 + j] = lg;
      unsigned bits = __float_as_uint(lg);
      unsigned key  = (bits & 0x80000000u) ? ~bits : (bits | 0x80000000u);
      atomicMax(&nmax[(size_t)sDst[e] * 8 + j], key);
    }
  }
}

// ---------------- exp + denominator ----------------
__global__ void exp_kernel(const int* __restrict__ ei, int E, float* __restrict__ exbuf,
                           const unsigned* __restrict__ nmax, float* __restrict__ den) {
  int idx = blockIdx.x * blockDim.x + threadIdx.x;
  if (idx >= E * 8) return;
  int e = idx >> 3, j = idx & 7;
  int d = ei[E + e];
  unsigned key  = nmax[(size_t)d * 8 + j];
  unsigned bits = (key & 0x80000000u) ? (key & 0x7FFFFFFFu) : ~key;
  float mx = __uint_as_float(bits);
  float ex = __expf(exbuf[idx] - mx);
  exbuf[idx] = ex;
  atomicAdd(&den[(size_t)d * 8 + j], ex);
}

// ---------------- msg, agg scatter, coordinate update ----------------
__global__ __launch_bounds__(128) void msg_kernel(
    const float* __restrict__ x, const int* __restrict__ ei, int E,
    const _Float16* __restrict__ vbuf, const float* __restrict__ exbuf,
    const float* __restrict__ den,
    const _Float16* __restrict__ xm_w1h, const float* __restrict__ xm_b1,
    const float* __restrict__ xm_w2,
    float* __restrict__ agg, float* __restrict__ dxbuf) {
  __shared__ __align__(16) _Float16 sM[TILE_E * 128];
  __shared__ __align__(16) _Float16 sTh[TILE_E * 128];
  __shared__ float sAttn[TILE_E * 8];
  __shared__ int sDst[TILE_E], sVal[TILE_E];

  const int tid = threadIdx.x, wave = tid >> 5, lane = tid & 31;
  const int base = blockIdx.x * TILE_E;

  if (tid < TILE_E) {
    int e = base + tid;
    int valid = (e < E);
    int d = valid ? ei[E + e] : 0;
    sDst[tid] = d; sVal[tid] = valid;
    for (int j = 0; j < 8; ++j) {
      float a = 0.f;
      if (valid) a = exbuf[(size_t)e * 8 + j] / (den[(size_t)d * 8 + j] + 1e-16f);
      sAttn[tid * 8 + j] = a;
    }
  }
  __syncthreads();

  // msg = attn * v ; scatter into agg
  {
    int eL = tid >> 1, off = (tid & 1) << 6;
    bool valid = sVal[eL] != 0;
    int d = sDst[eL];
    int ge = base + eL;
    for (int c = 0; c < 64; ++c) {
      int col = off + c;
      float mv = valid ? sAttn[eL * 8 + (col >> 4)] * (float)vbuf[(size_t)ge * 128 + col] : 0.f;
      sM[eL * 128 + col] = (_Float16)mv;
      if (valid) atomicAdd(&agg[(size_t)d * 128 + col], mv);
    }
  }
  __syncthreads();

  // msg @ xm_w1 -> silu -> f16
  {
    const _Float16* Am = sM + wave * 16 * 128;
    const int g = lane >> 4, cl = lane & 15;
    v16h a[4];
#pragma unroll
    for (int kk = 0; kk < 4; ++kk) a[kk] = frag_a(Am + kk * 32, 128);
    for (int nt = 0; nt < 8; ++nt) {
      v8f c = {};
#pragma unroll
      for (int kk = 0; kk < 4; ++kk)
        c = wmma32(a[kk], frag_b_sw(xm_w1h + (size_t)(kk * 8 + nt) * 512), c);
      int col = nt * 16 + cl;
      float b = xm_b1[col];
#pragma unroll
      for (int r = 0; r < 8; ++r) {
        float t = c[r] + b;
        float s = t / (1.f + __expf(-t));
        sTh[(wave * 16 + r + 8 * g) * 128 + col] = (_Float16)s;
      }
    }
  }
  __syncthreads();

  if (tid < TILE_E && sVal[tid]) {
    int e = base + tid;
    float acc = 0.f;
    for (int c = 0; c < 128; ++c) acc += (float)sTh[tid * 128 + c] * xm_w2[c];
    float coef = tanhf(acc);
    int s = ei[e], d = sDst[tid];
    float rx = x[d * 3 + 0] - x[s * 3 + 0];
    float ry = x[d * 3 + 1] - x[s * 3 + 1];
    float rz = x[d * 3 + 2] - x[s * 3 + 2];
    float dist = sqrtf(rx * rx + ry * ry + rz * rz + 1e-8f);
    float f = coef / (dist + 1.f);
    atomicAdd(&dxbuf[(size_t)d * 3 + 0], rx * f);
    atomicAdd(&dxbuf[(size_t)d * 3 + 1], ry * f);
    atomicAdd(&dxbuf[(size_t)d * 3 + 2], rz * f);
  }
}

// ---------------- node update ----------------
__global__ __launch_bounds__(128) void node_kernel(
    const float* __restrict__ h, const float* __restrict__ x, int N,
    const _Float16* __restrict__ hf, const float* __restrict__ agg,
    const _Float16* __restrict__ out_wh, const float* __restrict__ out_b,
    const _Float16* __restrict__ nm_w1h, const float* __restrict__ nm_b1,
    const float* __restrict__ nm_g, const float* __restrict__ nm_be,
    const _Float16* __restrict__ nm_w2h, const float* __restrict__ nm_b2,
    const float* __restrict__ dxbuf, const unsigned char* __restrict__ mask,
    float* __restrict__ out) {
  __shared__ __align__(16) _Float16 sAh[TILE_N * 128];  // agg f16 / reused for LN out
  __shared__ __align__(16) _Float16 sA2[TILE_N * 256];  // [aggW | h]
  __shared__ __align__(16) float    sY32[TILE_N * 128];

  const int tid = threadIdx.x, wave = tid >> 5, lane = tid & 31;
  const int base = blockIdx.x * TILE_N;

  {
    int nL = tid >> 1, off = (tid & 1) << 6;
    int node = base + nL;
    bool valid = node < N;
    for (int c = 0; c < 64; ++c) {
      int col = off + c;
      sAh[nL * 128 + col] = valid ? (_Float16)agg[(size_t)node * 128 + col] : (_Float16)0.f;
      sA2[nL * 256 + 128 + col] = valid ? hf[(size_t)node * 128 + col] : (_Float16)0.f;
    }
  }
  __syncthreads();

  // aggW = agg @ out_w + out_b  -> sA2 cols 0..127 (f16)
  {
    const _Float16* Aa = sAh + wave * 16 * 128;
    const int g = lane >> 4, cl = lane & 15;
    v16h a[4];
#pragma unroll
    for (int kk = 0; kk < 4; ++kk) a[kk] = frag_a(Aa + kk * 32, 128);
    for (int nt = 0; nt < 8; ++nt) {
      v8f c = {};
#pragma unroll
      for (int kk = 0; kk < 4; ++kk)
        c = wmma32(a[kk], frag_b_sw(out_wh + (size_t)(kk * 8 + nt) * 512), c);
      int col = nt * 16 + cl;
      float b = out_b[col];
#pragma unroll
      for (int r = 0; r < 8; ++r)
        sA2[(wave * 16 + r + 8 * g) * 256 + col] = (_Float16)(c[r] + b);
    }
  }
  __syncthreads();

  // [aggW|h] @ nm_w1 (K=256) -> sY32
  {
    const _Float16* Aa = sA2 + wave * 16 * 256;
    const int g = lane >> 4, cl = lane & 15;
    v16h a[8];
#pragma unroll
    for (int kk = 0; kk < 8; ++kk) a[kk] = frag_a(Aa + kk * 32, 256);
    for (int nt = 0; nt < 8; ++nt) {
      v8f c = {};
#pragma unroll
      for (int kk = 0; kk < 8; ++kk)
        c = wmma32(a[kk], frag_b_sw(nm_w1h + (size_t)(kk * 8 + nt) * 512), c);
      int col = nt * 16 + cl;
      float b = nm_b1[col];
#pragma unroll
      for (int r = 0; r < 8; ++r)
        sY32[(wave * 16 + r + 8 * g) * 128 + col] = c[r] + b;
    }
  }
  __syncthreads();
  if (tid < TILE_N) ln_relu_row(sY32 + tid * 128, nm_g, nm_be, sAh + tid * 128);
  __syncthreads();

  // @ nm_w2 + nm_b2 + residual -> h_out
  {
    const _Float16* Aa = sAh + wave * 16 * 128;
    const int g = lane >> 4, cl = lane & 15;
    v16h a[4];
#pragma unroll
    for (int kk = 0; kk < 4; ++kk) a[kk] = frag_a(Aa + kk * 32, 128);
    for (int nt = 0; nt < 8; ++nt) {
      v8f c = {};
#pragma unroll
      for (int kk = 0; kk < 4; ++kk)
        c = wmma32(a[kk], frag_b_sw(nm_w2h + (size_t)(kk * 8 + nt) * 512), c);
      int col = nt * 16 + cl;
      float b = nm_b2[col];
#pragma unroll
      for (int r = 0; r < 8; ++r) {
        int node = base + wave * 16 + r + 8 * g;
        if (node < N)
          out[(size_t)node * 128 + col] = h[(size_t)node * 128 + col] + c[r] + b;
      }
    }
  }

  // x_out = x + dx * mask
  if (tid < TILE_N) {
    int node = base + tid;
    if (node < N) {
      float mv = mask[node] ? 1.f : 0.f;
      size_t xo = (size_t)N * 128;
      for (int k = 0; k < 3; ++k)
        out[xo + (size_t)node * 3 + k] = x[node * 3 + k] + dxbuf[(size_t)node * 3 + k] * mv;
    }
  }
}

// ---------------- host launcher ----------------
extern "C" void kernel_launch(void* const* d_in, const int* in_sizes, int n_in,
                              void* d_out, int out_size, void* d_ws, size_t ws_size,
                              hipStream_t stream) {
  (void)n_in; (void)out_size; (void)ws_size;
  const float* h  = (const float*)d_in[0];
  const float* x  = (const float*)d_in[1];
  const float* ea = (const float*)d_in[2];
  const float* kv_w1 = (const float*)d_in[3];
  const float* kv_b1 = (const float*)d_in[4];
  const float* kv_g  = (const float*)d_in[5];
  const float* kv_be = (const float*)d_in[6];
  const float* kv_w2 = (const float*)d_in[7];
  const float* kv_b2 = (const float*)d_in[8];
  const float* q_w   = (const float*)d_in[9];
  const float* q_b   = (const float*)d_in[10];
  const float* ab_w1 = (const float*)d_in[11];
  const float* ab_b1 = (const float*)d_in[12];
  const float* ab_g  = (const float*)d_in[13];
  const float* ab_be = (const float*)d_in[14];
  const float* ab_w2 = (const float*)d_in[15];
  const float* ab_b2 = (const float*)d_in[16];
  const float* out_w = (const float*)d_in[17];
  const float* out_b = (const float*)d_in[18];
  const float* nm_w1 = (const float*)d_in[19];
  const float* nm_b1 = (const float*)d_in[20];
  const float* nm_g  = (const float*)d_in[21];
  const float* nm_be = (const float*)d_in[22];
  const float* nm_w2 = (const float*)d_in[23];
  const float* nm_b2 = (const float*)d_in[24];
  const float* xm_w1 = (const float*)d_in[25];
  const float* xm_b1 = (const float*)d_in[26];
  const float* xm_w2 = (const float*)d_in[27];
  const int*   eidx  = (const int*)d_in[28];
  const unsigned char* mask = (const unsigned char*)d_in[29];

  const int N = in_sizes[0] / 128;
  const int E = in_sizes[28] / 2;

  // carve workspace
  char* ws = (char*)d_ws;
  size_t off = 0;
  auto carve = [&](size_t bytes) -> void* {
    void* p = ws + off;
    off += (bytes + 255) & ~(size_t)255;
    return p;
  };
  _Float16* kv_w1h = (_Float16*)carve((size_t)192 * 128 * 2);
  _Float16* kv_w2h = (_Float16*)carve((size_t)128 * 256 * 2);
  _Float16* q_wh   = (_Float16*)carve((size_t)128 * 128 * 2);
  _Float16* ab_w1h = (_Float16*)carve((size_t)64 * 128 * 2);
  _Float16* out_wh = (_Float16*)carve((size_t)128 * 128 * 2);
  _Float16* nm_w1h = (_Float16*)carve((size_t)256 * 128 * 2);
  _Float16* nm_w2h = (_Float16*)carve((size_t)128 * 128 * 2);
  _Float16* xm_w1h = (_Float16*)carve((size_t)128 * 128 * 2);
  _Float16* hf     = (_Float16*)carve((size_t)N * 128 * 2);
  _Float16* vbuf   = (_Float16*)carve((size_t)E * 128 * 2);
  float*    exbuf  = (float*)carve((size_t)E * 8 * 4);
  unsigned* nmax   = (unsigned*)carve((size_t)N * 8 * 4);
  float*    den    = (float*)carve((size_t)N * 8 * 4);
  float*    agg    = (float*)carve((size_t)N * 128 * 4);
  float*    dxbuf  = (float*)carve((size_t)N * 3 * 4);

  auto sw = [&](const float* w, _Float16* o, int K, int Nn) {
    int n = K * Nn;
    swizzle_b_kernel<<<(n + 255) / 256, 256, 0, stream>>>(w, o, K, Nn);
  };
  sw(kv_w1, kv_w1h, 192, 128);
  sw(kv_w2, kv_w2h, 128, 256);
  sw(q_w,   q_wh,   128, 128);
  sw(ab_w1, ab_w1h, 64, 128);
  sw(out_w, out_wh, 128, 128);
  sw(nm_w1, nm_w1h, 256, 128);
  sw(nm_w2, nm_w2h, 128, 128);
  sw(xm_w1, xm_w1h, 128, 128);
  cast_f16_kernel<<<(N * 128 + 255) / 256, 256, 0, stream>>>(h, hf, N * 128);

  hipMemsetAsync(nmax, 0, (size_t)N * 8 * 4, stream);
  hipMemsetAsync(den,  0, (size_t)N * 8 * 4, stream);
  hipMemsetAsync(agg,  0, (size_t)N * 128 * 4, stream);
  hipMemsetAsync(dxbuf, 0, (size_t)N * 3 * 4, stream);

  const int gridE = (E + TILE_E - 1) / TILE_E;
  edge_main_kernel<<<gridE, 128, 0, stream>>>(
      x, ea, eidx, E, hf, kv_w1h, kv_b1, kv_g, kv_be, kv_w2h, kv_b2, q_wh, q_b,
      ab_w1h, ab_b1, ab_g, ab_be, ab_w2, ab_b2, vbuf, exbuf, nmax);
  exp_kernel<<<(E * 8 + 255) / 256, 256, 0, stream>>>(eidx, E, exbuf, nmax, den);
  msg_kernel<<<gridE, 128, 0, stream>>>(x, eidx, E, vbuf, exbuf, den,
                                        xm_w1h, xm_b1, xm_w2, agg, dxbuf);
  node_kernel<<<(N + TILE_N - 1) / TILE_N, 128, 0, stream>>>(
      h, x, N, hf, agg, out_wh, out_b, nm_w1h, nm_b1, nm_g, nm_be, nm_w2h, nm_b2,
      dxbuf, mask, (float*)d_out);
}
